// TemporalAttention_77060303224945
// MI455X (gfx1250) — matrix operand order repaired and checked by reading, not compiled, not run on hardware
//
#include <hip/hip_runtime.h>
#include <math.h>

// ---- problem constants (from reference) ----
#define NHEADS    8
#define NODE_DIM  172
#define EDGE_DIM  172
#define TIME_DIM  100
#define OUT_DIM   272     // = 17*16
#define HEAD_DIM  34
#define KEY_DIM   444
#define KEY_PAD   448     // padded to 14 * 32 for bf16 WMMA k-loop
#define KV_DIM    544     // = 34*16 (K then V)
#define NBR       64
#define BATCH     4096
#define LN_EPS    1e-5f

// LDS strides chosen for bank behavior / alignment
#define ZIN_STRIDE16 456  // bf16 units; 912 B/row, 16B-aligned fragments
#define KV_STRIDE16  552  // bf16 units; 1104 B/row

typedef float  v2f  __attribute__((ext_vector_type(2)));
typedef float  v4f  __attribute__((ext_vector_type(4)));
typedef float  v8f  __attribute__((ext_vector_type(8)));
typedef __bf16 v4bf __attribute__((ext_vector_type(4)));
typedef __bf16 v8bf __attribute__((ext_vector_type(8)));
typedef __bf16 v16bf __attribute__((ext_vector_type(16)));

#if __has_builtin(__builtin_amdgcn_global_load_async_to_lds_b32) && \
    __has_builtin(__builtin_amdgcn_s_wait_asynccnt)
#define USE_ASYNC_LDS 1
typedef __attribute__((address_space(1))) int gas_int;   // global addrspace int
typedef __attribute__((address_space(3))) int las_int;   // LDS addrspace int
#else
#define USE_ASYNC_LDS 0
#endif

// ---- pre-kernel: W_KV (544x444 f32) -> bf16, zero-padded to 544x448 ----
__global__ void convert_wkv_kernel(const float* __restrict__ W_KV,
                                   __bf16* __restrict__ W_KVb)
{
    const int idx = blockIdx.x * 256 + threadIdx.x;
    if (idx >= KV_DIM * KEY_PAD) return;
    const int n = idx / KEY_PAD, k = idx % KEY_PAD;
    W_KVb[idx] = (k < KEY_DIM) ? (__bf16)W_KV[(long)n * KEY_DIM + k] : (__bf16)0.f;
}

__launch_bounds__(256)
__global__ void tga_fused_kernel(const float* __restrict__ node_feat,
                                 const float* __restrict__ time_feat,
                                 const float* __restrict__ edge_feat,
                                 const float* __restrict__ nbr_node_feat,
                                 const float* __restrict__ nbr_time_feat,
                                 const int*   __restrict__ nbr_mask,
                                 const float* __restrict__ W_Q,
                                 const __bf16* __restrict__ W_KVb,
                                 const float* __restrict__ W_O,
                                 const float* __restrict__ b_O,
                                 const float* __restrict__ ln_gamma,
                                 const float* __restrict__ ln_beta,
                                 float* __restrict__ out)
{
    extern __shared__ float smem[];
    float*  r   = smem;                       // 272
    float*  q   = r   + OUT_DIM;              // 272
    float*  sc  = q   + OUT_DIM;              // 8*64 scores -> probs
    float*  o   = sc  + NHEADS * NBR;         // 272
    float*  xb  = o   + OUT_DIM;              // 272
    float*  red = xb  + OUT_DIM;              // 32  (float region = 1632 floats, 16B multiple)
    __bf16* kv16  = (__bf16*)(red + 32);      // 64 * 552 bf16 (K cols 0..271, V cols 272..543)
    __bf16* zin16 = kv16 + NBR * KV_STRIDE16; // 64 * 456 bf16
    // total LDS = 6528 + 70656 + 58368 = 135,552 B -> 2 blocks / 320 KB WGP

    const int b    = blockIdx.x;
    const int tid  = threadIdx.x;
    const int wave = tid >> 5;
    const int lane = tid & 31;
    const int l15  = lane & 15;
    const int ksel = lane >> 4;               // 0 for lanes 0-15, 1 for lanes 16-31

    // ---- Phase 0: R = concat(node_feat[b], time_feat[b,0,:]) into LDS ----
#if USE_ASYNC_LDS
    if (tid < OUT_DIM) {
        const float* src = (tid < NODE_DIM)
                         ? node_feat + (long)b * NODE_DIM + tid
                         : time_feat + (long)b * TIME_DIM + (tid - NODE_DIM);
        __builtin_amdgcn_global_load_async_to_lds_b32(
            (gas_int*)src, (las_int*)(r + tid), 0, 0);
    }
#else
    for (int d = tid; d < OUT_DIM; d += 256) {
        r[d] = (d < NODE_DIM) ? node_feat[(long)b * NODE_DIM + d]
                              : time_feat[(long)b * TIME_DIM + (d - NODE_DIM)];
    }
#endif

    // ---- Phase 0b: stage Z_in -> LDS as bf16 (444 f32/row = 111 float4 chunks) ----
    const long nb = (long)b * NBR;
    for (int idx = tid; idx < NBR * 111; idx += 256) {
        const int row = idx / 111, j = idx % 111;
        v4f val; int col;
        if (j < 43)      { col = j * 4;
                           val = *(const v4f*)(nbr_node_feat + (nb + row) * NODE_DIM + col); }
        else if (j < 86) { col = NODE_DIM + (j - 43) * 4;
                           val = *(const v4f*)(edge_feat     + (nb + row) * EDGE_DIM + (j - 43) * 4); }
        else             { col = NODE_DIM + EDGE_DIM + (j - 86) * 4;
                           val = *(const v4f*)(nbr_time_feat + (nb + row) * TIME_DIM + (j - 86) * 4); }
        v4bf ov = { (__bf16)val.x, (__bf16)val.y, (__bf16)val.z, (__bf16)val.w };
        *(v4bf*)(zin16 + row * ZIN_STRIDE16 + col) = ov;
    }
    // zero pad cols 444..447 (64 rows * 4 cols == 256 threads, exactly)
    zin16[(tid >> 2) * ZIN_STRIDE16 + KEY_DIM + (tid & 3)] = (__bf16)0.f;

#if USE_ASYNC_LDS
    __builtin_amdgcn_s_wait_asynccnt(0);
#endif
    __syncthreads();

    // ---- Phase 1: Q = R @ W_Q^T (tiny; W_Q is L2-hot) ----
    for (int d = tid; d < OUT_DIM; d += 256) {
        const float* w = W_Q + (long)d * OUT_DIM;
        float acc = 0.f;
        for (int e = 0; e < OUT_DIM; ++e) acc = fmaf(r[e], w[e], acc);
        q[d] = acc;
    }

    // ---- Phase 2: KV = Z_in @ W_KV^T via V_WMMA_F32_16X16X32_BF16 ----
    // Work unit = one N-tile (16 kv dims) x all 4 M-tiles (64 nbrs): B-frag reused 4x.
    for (int nt = wave; nt < KV_DIM / 16; nt += 8) {
        const int n0 = nt * 16;
        const __bf16* wrow = W_KVb + (long)(n0 + l15) * KEY_PAD + ksel * 8;
        const __bf16* zcol = zin16 + l15 * ZIN_STRIDE16 + ksel * 8;
        v8f acc[4] = {{}, {}, {}, {}};
        for (int k = 0; k < KEY_PAD; k += 32) {
            v8bf blo = *(const v8bf*)(wrow + k);        // B: K = k + 8*ksel .. +7 for col n0+l15
            v8bf bhi = *(const v8bf*)(wrow + k + 16);   // K = k + 16 + 8*ksel .. +7
            v16bf bfrag = __builtin_shufflevector(blo, bhi, 0,1,2,3,4,5,6,7,8,9,10,11,12,13,14,15);
            #pragma unroll
            for (int mt = 0; mt < 4; ++mt) {
                const __bf16* zr = zcol + mt * 16 * ZIN_STRIDE16 + k;
                v8bf alo = *(const v8bf*)(zr);          // K = k + 8*ksel .. +7
                v8bf ahi = *(const v8bf*)(zr + 16);     // K = k + 16 + 8*ksel .. +7
                v16bf afrag = __builtin_shufflevector(alo, ahi,
                    0,1,2,3,4,5,6,7,8,9,10,11,12,13,14,15);
                acc[mt] = __builtin_amdgcn_wmma_f32_16x16x32_bf16(
                              false, afrag, false, bfrag, (short)0, acc[mt], false, false);
            }
        }
        #pragma unroll
        for (int mt = 0; mt < 4; ++mt) {
            // D layout: VGPR v -> row mt*16 + v + 8*ksel, col n0 + l15
            __bf16* drow = kv16 + (mt * 16 + ksel * 8) * KV_STRIDE16 + n0 + l15;
            #pragma unroll
            for (int v = 0; v < 8; ++v) drow[v * KV_STRIDE16] = (__bf16)acc[mt][v];
        }
    }
    __syncthreads();

    // ---- Phase 3: scores = (Qh . Kh) * HEAD_DIM^-0.5, masked ----
    const int* mask = nbr_mask + (long)b * NBR;
    const float scale = 0.17149858514250882f; // 1/sqrt(34)
    for (int p = tid; p < NHEADS * NBR; p += 256) {
        const int h = p >> 6, n = p & 63;
        const __bf16* kr = kv16 + n * KV_STRIDE16 + h * HEAD_DIM;
        const float* qh = q + h * HEAD_DIM;
        float s = 0.f;
        for (int d2 = 0; d2 < HEAD_DIM; ++d2) s = fmaf(qh[d2], (float)kr[d2], s);
        s *= scale;
        if (mask[n] == 0) s = -1e10f;
        sc[p] = s;
    }
    __syncthreads();

    // ---- Phase 4: softmax over 64 neighbors; wave w owns head w ----
    {
        const int h = wave;
        float v0 = sc[h * 64 + lane];
        float v1 = sc[h * 64 + 32 + lane];
        float m = fmaxf(v0, v1);
        #pragma unroll
        for (int off = 16; off > 0; off >>= 1) m = fmaxf(m, __shfl_xor(m, off, 32));
        float e0 = __expf(v0 - m), e1 = __expf(v1 - m);
        float s = e0 + e1;
        #pragma unroll
        for (int off = 16; off > 0; off >>= 1) s += __shfl_xor(s, off, 32);
        const float inv = 1.f / s;
        sc[h * 64 + lane]      = e0 * inv;
        sc[h * 64 + 32 + lane] = e1 * inv;
    }
    __syncthreads();

    // ---- Phase 5: O[d] = sum_n P[h][n] * V[n][d]  (V = kv16 cols 272..543) ----
    for (int d = tid; d < OUT_DIM; d += 256) {
        const int h = d / HEAD_DIM;
        const float* pr = sc + h * 64;
        const __bf16* vr = kv16 + OUT_DIM + d;
        float acc = 0.f;
        for (int n = 0; n < NBR; ++n) acc = fmaf(pr[n], (float)vr[n * KV_STRIDE16], acc);
        o[d] = acc;
    }
    __syncthreads();

    // ---- Phase 6: x = O @ W_O^T + b_O + R ----
    for (int d = tid; d < OUT_DIM; d += 256) {
        const float* w = W_O + (long)d * OUT_DIM;
        float acc = b_O[d] + r[d];
        for (int e = 0; e < OUT_DIM; ++e) acc = fmaf(o[e], w[e], acc);
        xb[d] = acc;
    }
    __syncthreads();

    // ---- Phase 7: LayerNorm over 272 dims ----
    float ls = 0.f, lq = 0.f;
    for (int d = tid; d < OUT_DIM; d += 256) { float x = xb[d]; ls += x; lq += x * x; }
    #pragma unroll
    for (int off = 16; off > 0; off >>= 1) {
        ls += __shfl_xor(ls, off, 32);
        lq += __shfl_xor(lq, off, 32);
    }
    if (lane == 0) { red[wave] = ls; red[8 + wave] = lq; }
    __syncthreads();
    if (tid == 0) {
        float s = 0.f, sq = 0.f;
        for (int wv = 0; wv < 8; ++wv) { s += red[wv]; sq += red[8 + wv]; }
        const float mu  = s / (float)OUT_DIM;
        const float var = sq / (float)OUT_DIM - mu * mu;
        red[16] = mu;
        red[17] = rsqrtf(var + LN_EPS);
    }
    __syncthreads();
    const float mu = red[16], rstd = red[17];
    for (int d = tid; d < OUT_DIM; d += 256) {
        out[(long)b * OUT_DIM + d] = (xb[d] - mu) * rstd * ln_gamma[d] + ln_beta[d];
    }
}

extern "C" void kernel_launch(void* const* d_in, const int* in_sizes, int n_in,
                              void* d_out, int out_size, void* d_ws, size_t ws_size,
                              hipStream_t stream) {
    const float* node_feat     = (const float*)d_in[0];
    const float* time_feat     = (const float*)d_in[1];
    const float* edge_feat     = (const float*)d_in[2];
    const float* nbr_node_feat = (const float*)d_in[3];
    const float* nbr_time_feat = (const float*)d_in[4];
    const int*   nbr_mask      = (const int*)  d_in[5];
    const float* W_Q           = (const float*)d_in[6];
    const float* W_KV          = (const float*)d_in[7];
    const float* W_O           = (const float*)d_in[8];
    const float* b_O           = (const float*)d_in[9];
    const float* ln_gamma      = (const float*)d_in[10];
    const float* ln_beta       = (const float*)d_in[11];
    float*  out   = (float*)d_out;
    __bf16* W_KVb = (__bf16*)d_ws;   // 544*448*2 = 487,424 B of scratch

    // Pre-pass: fp32 -> bf16 (zero-padded) copy of W_KV, rebuilt every call (deterministic).
    const int nconv = KV_DIM * KEY_PAD;
    convert_wkv_kernel<<<(nconv + 255) / 256, 256, 0, stream>>>(W_KV, W_KVb);

    const size_t shmem = (size_t)(4 * OUT_DIM + NHEADS * NBR + 32) * sizeof(float)   // r,q,o,xb,sc,red
                       + (size_t)(NBR * KV_STRIDE16) * sizeof(__bf16)                // kv16
                       + (size_t)(NBR * ZIN_STRIDE16) * sizeof(__bf16);              // zin16
                       // = 135,552 B -> 2 blocks per 320 KB WGP

    tga_fused_kernel<<<BATCH, 256, shmem, stream>>>(
        node_feat, time_feat, edge_feat, nbr_node_feat, nbr_time_feat, nbr_mask,
        W_Q, W_KVb, W_O, b_O, ln_gamma, ln_beta, out);
}
